// WindowAttention_48447231099307
// MI455X (gfx1250) — matrix operand order repaired and checked by reading, not compiled, not run on hardware
//
#include <hip/hip_runtime.h>

typedef __attribute__((ext_vector_type(16))) _Float16 v16h;
typedef __attribute__((ext_vector_type(8)))  _Float16 v8h;
typedef __attribute__((ext_vector_type(4)))  _Float16 v4h;
typedef __attribute__((ext_vector_type(8)))  float    v8f;
typedef int v4i_vs __attribute__((vector_size(16)));   // async-LDS builtin param type

#define NTOK   64
#define CDIM   192
#define NHEADS 6

#if defined(__HIP_DEVICE_COMPILE__) && __has_builtin(__builtin_amdgcn_global_load_async_to_lds_b128)
#define HAVE_ASYNC_LDS 1
#else
#define HAVE_ASYNC_LDS 0
#endif

// ---------------------------------------------------------------------------
// Weight prep: f32 -> f16, swizzled into WMMA B-fragment layout.
// B fragment (16-bit, 32x16 KxN): lane = (N&15) + 16*((K>>3)&1), e = (K&7) + 8*((K>>4)&1).
// 512-half blocks per (ntile, ktile), ktile-major inner.
// ---------------------------------------------------------------------------
__global__ void prep_weights_kernel(const float* __restrict__ qkv_w,
                                    const float* __restrict__ proj_w,
                                    _Float16* __restrict__ wq,
                                    _Float16* __restrict__ wp) {
    int i = blockIdx.x * 256 + threadIdx.x;
    if (i < 576 * 192) {
        int o = i / 192, c = i % 192;
        int nt = o >> 4, kt = c >> 5;
        int la = (o & 15) + 16 * ((c >> 3) & 1);
        int e  = (c & 7) + 8 * ((c >> 4) & 1);
        wq[(nt * 6 + kt) * 512 + la * 16 + e] = (_Float16)qkv_w[i];
    }
    if (i < 192 * 192) {
        int o = i / 192, c = i % 192;
        int nt = o >> 4, kt = c >> 5;
        int la = (o & 15) + 16 * ((c >> 3) & 1);
        int e  = (c & 7) + 8 * ((c >> 4) & 1);
        wp[(nt * 6 + kt) * 512 + la * 16 + e] = (_Float16)proj_w[i];
    }
}

// ---------------------------------------------------------------------------
// Fused window attention: one workgroup (8 wave32s) per window.
// A frag (16-bit 16x32): lane = row + 16*((K>>3)&1), e = (K&7) + 8*(K>>4)
// B frag (16-bit 32x16): lane = (N&15) + 16*((K>>3)&1), e = (K&7) + 8*(K>>4)
// C/D (f32 16x16):       row = r + 8*(lane>>4),      col = lane&15
// ---------------------------------------------------------------------------
__launch_bounds__(256)
__global__ void win_attn_kernel(const float* __restrict__ x,
                                const float* __restrict__ mask,
                                const float* __restrict__ qkv_b,
                                const float* __restrict__ proj_b,
                                const _Float16* __restrict__ wq,
                                const _Float16* __restrict__ wp,
                                float* __restrict__ out) {
    // 128 KB LDS -> 2 workgroups per 320 KB WGP pool.
    __shared__ __align__(32) _Float16 xA[12288];     // x A-frags; reused for attn-out A-frags
    __shared__ __align__(32) _Float16 qA[12288];     // q A-frags (head, mtile)
    __shared__ __align__(32) _Float16 kB[12288];     // k B-frags (head, ntile)
    __shared__ __align__(32) _Float16 vB[12288];     // v B-frags (head, ktile, ntile)
    __shared__ __align__(32) float maskSw[NTOK * NTOK]; // mask, swizzled [(mt,nt)][lane][r]
    __shared__ __align__(32) _Float16 pScr[8 * 1024];   // mask staging, then P scratch

    const int tid  = threadIdx.x;
    const int wave = __builtin_amdgcn_readfirstlane(tid >> 5);
    const int lane = tid & 31;
    const int half = lane >> 4;
    const int lcol = lane & 15;
    const int b    = blockIdx.x;

    // ---- Stage 1: stage x (swizzled A-frags, f32->f16); mask -> LDS -------
    const float* xg = x + (size_t)b * (NTOK * CDIM);
#pragma unroll
    for (int t = 0; t < 12; ++t) {
        int idx = (tid + 256 * t) * 4;               // 4 consecutive channels
        float4 xv = *(const float4*)(xg + idx);
        int n = idx / CDIM, c = idx % CDIM;          // c % 4 == 0
        int mt = n >> 4, kt = c >> 5;
        int la = (n & 15) + 16 * ((c >> 3) & 1);
        int e  = (c & 7) + 8 * ((c >> 4) & 1);       // e in {0,4,8,12}
        v4h hv = { (_Float16)xv.x, (_Float16)xv.y, (_Float16)xv.z, (_Float16)xv.w };
        *(v4h*)&xA[(mt * 6 + kt) * 512 + la * 16 + e] = hv;
    }
    const float* mg = mask + (size_t)(b & 4095) * (NTOK * NTOK);
#if HAVE_ASYNC_LDS
    {   // async copy raw mask into staging (pScr), then swizzle LDS->LDS
        float* stg = (float*)pScr;
#pragma unroll
        for (int t = 0; t < 4; ++t) {
            int idx = (tid + 256 * t) * 4;
            __builtin_amdgcn_global_load_async_to_lds_b128(
                (v4i_vs*)(void*)(mg + idx), (v4i_vs*)(void*)(stg + idx), 0, 0);
        }
#if __has_builtin(__builtin_amdgcn_s_wait_asynccnt)
        __builtin_amdgcn_s_wait_asynccnt(0);
#else
        asm volatile("s_wait_asynccnt 0" ::: "memory");
#endif
        __syncthreads();
#pragma unroll
        for (int t = 0; t < 4; ++t) {
            int idx = (tid + 256 * t) * 4;
            float4 mv = *(const float4*)(stg + idx);
            int n = idx >> 6, m0 = idx & 63;
            int mt = n >> 4, nl = n & 15;
            int base = ((mt * 4 + (m0 >> 4)) * 32 + (m0 & 15) + 16 * (nl >> 3)) * 8 + (nl & 7);
            maskSw[base +  0] = mv.x;
            maskSw[base +  8] = mv.y;
            maskSw[base + 16] = mv.z;
            maskSw[base + 24] = mv.w;
        }
    }
#else
#pragma unroll
    for (int t = 0; t < 4; ++t) {
        int idx = (tid + 256 * t) * 4;
        float4 mv = *(const float4*)(mg + idx);
        int n = idx >> 6, m0 = idx & 63;
        int mt = n >> 4, nl = n & 15;
        int base = ((mt * 4 + (m0 >> 4)) * 32 + (m0 & 15) + 16 * (nl >> 3)) * 8 + (nl & 7);
        maskSw[base +  0] = mv.x;
        maskSw[base +  8] = mv.y;
        maskSw[base + 16] = mv.z;
        maskSw[base + 24] = mv.w;
    }
#endif
    __syncthreads();

    const float scale = 0.17677669529663687f;        // 1/sqrt(32)

    // ---- Stage 2: qkv projection (4x36 tiles, 18/wave) --------------------
    for (int it = 0; it < 18; ++it) {
        int t   = wave + 8 * it;
        int mt  = t / 36, nto = t % 36;
        v8f acc = {};
#pragma unroll
        for (int kt = 0; kt < 6; ++kt) {
            v16h a  = *(const v16h*)&xA[(mt * 6 + kt) * 512 + lane * 16];
            v16h bf = *(const v16h*)&wq[(size_t)(nto * 6 + kt) * 512 + lane * 16];
            acc = __builtin_amdgcn_wmma_f32_16x16x32_f16(false, a, false, bf,
                                                         (short)0, acc, false, false);
        }
        float bias = qkv_b[nto * 16 + lcol];
        int s = nto / 12;                            // wave-uniform: 0=q 1=k 2=v
        int h = (nto >> 1) % 6;                      // wave-uniform head
        int d = ((nto & 1) << 4) + lcol;
        if (s == 0) {                                // q -> A-frag, pre-scaled
            _Float16* dst = &qA[(h * 4 + mt) * 512
                                + (8 * half + 16 * ((d >> 3) & 1)) * 16
                                + (d & 7) + 8 * (d >> 4)];
#pragma unroll
            for (int r = 0; r < 8; ++r)
                dst[r * 16] = (_Float16)((acc[r] + bias) * scale);
        } else if (s == 1) {                         // k -> B-frag (N=key row, K=d)
            _Float16* dst = &kB[(h * 4 + mt) * 512
                                + (8 * half + 16 * ((d >> 3) & 1)) * 16
                                + (d & 7) + 8 * (d >> 4)];
#pragma unroll
            for (int r = 0; r < 8; ++r)
                dst[r * 16] = (_Float16)(acc[r] + bias);
        } else {                                     // v -> B-frag; 8 consecutive halves
            v8h hv;
#pragma unroll
            for (int r = 0; r < 8; ++r)
                hv[r] = (_Float16)(acc[r] + bias);
            *(v8h*)&vB[((h * 2 + (mt >> 1)) * 2 + (d >> 4)) * 512
                       + ((d & 15) + 16 * half) * 16 + 8 * (mt & 1)] = hv;
        }
    }
    __syncthreads();

    // ---- Stage 3: attention per (head, query row-tile); 3 tasks/wave ------
    _Float16* myP = &pScr[wave * 1024];
    for (int it = 0; it < 3; ++it) {
        int task = wave + 8 * it;
        int h = task >> 2, mt = task & 3;
        v16h aq = *(const v16h*)&qA[(h * 4 + mt) * 512 + lane * 16];
        v8f sacc[4];
#pragma unroll
        for (int nt = 0; nt < 4; ++nt) {             // S = q k^T (K=32, 1 WMMA)
            v8f z = {};
            v16h bk = *(const v16h*)&kB[(h * 4 + nt) * 512 + lane * 16];
            sacc[nt] = __builtin_amdgcn_wmma_f32_16x16x32_f16(false, aq, false, bk,
                                                              (short)0, z, false, false);
        }
#pragma unroll
        for (int nt = 0; nt < 4; ++nt) {             // + mask (2 x b128 per nt)
            const float* mp = &maskSw[((mt * 4 + nt) * 32 + lane) * 8];
            float4 m0 = *(const float4*)mp;
            float4 m1 = *(const float4*)(mp + 4);
            sacc[nt][0] += m0.x; sacc[nt][1] += m0.y; sacc[nt][2] += m0.z; sacc[nt][3] += m0.w;
            sacc[nt][4] += m1.x; sacc[nt][5] += m1.y; sacc[nt][6] += m1.z; sacc[nt][7] += m1.w;
        }
        // f32 softmax; each row lives in one 16-lane half-group.
        // Store UNNORMALIZED exp in P; apply 1/sum to the PV output rows.
        float inv[8];
#pragma unroll
        for (int r = 0; r < 8; ++r) {
            float mx = fmaxf(fmaxf(sacc[0][r], sacc[1][r]),
                             fmaxf(sacc[2][r], sacc[3][r]));
#pragma unroll
            for (int off = 1; off < 16; off <<= 1)
                mx = fmaxf(mx, __shfl_xor(mx, off, 32));
            float sum = 0.f;
#pragma unroll
            for (int nt = 0; nt < 4; ++nt) {
                float ev = __expf(sacc[nt][r] - mx);
                sacc[nt][r] = ev;
                sum += ev;
            }
#pragma unroll
            for (int off = 1; off < 16; off <<= 1)
                sum += __shfl_xor(sum, off, 32);
            inv[r] = 1.f / sum;
#pragma unroll
            for (int nt = 0; nt < 4; ++nt)
                myP[(r + 8 * half) * 64 + nt * 16 + lcol] = (_Float16)sacc[nt][r];
        }
        // Re-fragment P: A-frag halves are 2 contiguous 8-half runs per lane
        v16h ap[2];
#pragma unroll
        for (int kt = 0; kt < 2; ++kt) {
            v8h lo = *(const v8h*)&myP[lcol * 64 + kt * 32 + 8 * half];       // K=0..7  (+8*half)
            v8h hi = *(const v8h*)&myP[lcol * 64 + kt * 32 + 16 + 8 * half];  // K=16..23(+8*half)
            ap[kt] = __builtin_shufflevector(lo, hi, 0, 1, 2, 3, 4, 5, 6, 7,
                                             8, 9, 10, 11, 12, 13, 14, 15);
        }
        // O = P v (K=64, 2 steps) -> proj A-frags (c = h*32+d, ktile == h)
#pragma unroll
        for (int ntd = 0; ntd < 2; ++ntd) {
            v8f oacc = {};
#pragma unroll
            for (int kt = 0; kt < 2; ++kt) {
                v16h bv = *(const v16h*)&vB[((h * 2 + kt) * 2 + ntd) * 512 + lane * 16];
                oacc = __builtin_amdgcn_wmma_f32_16x16x32_f16(false, ap[kt], false, bv,
                                                              (short)0, oacc, false, false);
            }
            int d = ntd * 16 + lcol;
            _Float16* dst = &xA[(mt * 6 + h) * 512
                                + (8 * half + 16 * ((d >> 3) & 1)) * 16
                                + (d & 7) + 8 * (d >> 4)];
#pragma unroll
            for (int r = 0; r < 8; ++r)
                dst[r * 16] = (_Float16)(oacc[r] * inv[r]);   // fold softmax norm here
        }
    }
    __syncthreads();

    // ---- Stage 4: output projection, 48 tiles (6/wave) --------------------
    float* og = out + (size_t)b * (NTOK * CDIM);
    for (int it = 0; it < 6; ++it) {
        int t  = wave + 8 * it;
        int mt = t / 12, nt = t % 12;
        v8f acc = {};
#pragma unroll
        for (int kt = 0; kt < 6; ++kt) {
            v16h a  = *(const v16h*)&xA[(mt * 6 + kt) * 512 + lane * 16];
            v16h bw = *(const v16h*)&wp[(size_t)(nt * 6 + kt) * 512 + lane * 16];
            acc = __builtin_amdgcn_wmma_f32_16x16x32_f16(false, a, false, bw,
                                                         (short)0, acc, false, false);
        }
        int o = nt * 16 + lcol;
        float bias = proj_b[o];
#pragma unroll
        for (int r = 0; r < 8; ++r) {
            int n = mt * 16 + r + 8 * half;
            og[n * CDIM + o] = acc[r] + bias;        // lanes 0..15 -> consecutive o
        }
    }
}

extern "C" void kernel_launch(void* const* d_in, const int* in_sizes, int n_in,
                              void* d_out, int out_size, void* d_ws, size_t ws_size,
                              hipStream_t stream) {
    const float* x      = (const float*)d_in[0];
    const float* mask   = (const float*)d_in[1];
    const float* qkv_w  = (const float*)d_in[2];
    const float* qkv_b  = (const float*)d_in[3];
    const float* proj_w = (const float*)d_in[4];
    const float* proj_b = (const float*)d_in[5];

    _Float16* wq = (_Float16*)d_ws;          // 110592 halves (216 KB)
    _Float16* wp = wq + 110592;              //  36864 halves ( 72 KB)

    prep_weights_kernel<<<(576 * 192 + 255) / 256, 256, 0, stream>>>(qkv_w, proj_w, wq, wp);
    win_attn_kernel<<<8192, 256, 0, stream>>>(x, mask, qkv_b, proj_b, wq, wp,
                                              (float*)d_out);
}
